// Cross_WindowAttention_3238405341249
// MI455X (gfx1250) — matrix operand order repaired and checked by reading, not compiled, not run on hardware
//
#include <hip/hip_runtime.h>
#include <hip/hip_bf16.h>

// Cross window attention (Swin-style), fused single-pass per window.
// One workgroup (256 thr = 8 wave32) per window; all GEMMs via
// v_wmma_f32_16x16x32_f16; intermediates live entirely in LDS (144 KB).

typedef _Float16 h16;
typedef __attribute__((ext_vector_type(16))) _Float16 v16h;
typedef __attribute__((ext_vector_type(8)))  _Float16 v8h;
typedef __attribute__((ext_vector_type(8)))  float    v8f;

#define NTOK  49
#define NPAD  64
#define DIM   96
#define HEADS 3
#define HD    32
#define KVDIM 192
#define ATTN_SCALE 0.17677669529663687f   // 32^-0.5

// ---------------- WMMA fragment loaders (ISA 7.12.2 layouts) ----------------

// A (16x32 f16): lane -> M = lane&15; halves i cover K with 8/16 interleave:
//   K = i + (i>=8 ? 8 : 0) + ((lane>>4)<<3)
__device__ __forceinline__ v16h load_a_frag(const h16* base, int ld,
                                            int mBase, int kBase, int lane) {
  const h16* p = base + (mBase + (lane & 15)) * ld + kBase + ((lane >> 4) << 3);
  v16h a;
#pragma unroll
  for (int i = 0; i < 8; ++i)  a[i] = p[i];
#pragma unroll
  for (int i = 8; i < 16; ++i) a[i] = p[i + 8];
  return a;
}

// B (32x16 f16), row-major K x N source: lane -> N = lane&15; halves i cover K
// with 8/16 interleave: K = i + (i>=8 ? 8 : 0) + ((lane>>4)<<3) (strided gather).
__device__ __forceinline__ v16h load_b_frag_vec(const h16* base, int ld,
                                                int kBase, int nBase, int lane) {
  const h16* p = base + (size_t)(kBase + ((lane >> 4) << 3)) * ld + nBase + (lane & 15);
  v16h b;
#pragma unroll
  for (int i = 0; i < 8; ++i)  b[i] = p[(size_t)i * ld];
#pragma unroll
  for (int i = 8; i < 16; ++i) b[i] = p[(size_t)(i + 8) * ld];
  return b;
}

__device__ __forceinline__ v8f wmma_f16(v16h a, v16h b, v8f c) {
  return __builtin_amdgcn_wmma_f32_16x16x32_f16(false, a, false, b,
                                                (short)0, c, false, false);
}

// ---------------- one-time weight transpose + f16 convert ----------------
// qwT [96][96]  : qwT[ci*96+co]  = q_w[co*96+ci]
// kvwT[96][192] : kvwT[ci*192+co]= kv_w[co*96+ci]
// pwT [96][96]  : pwT[ci*96+co]  = proj_w[co*96+ci]
__global__ void prep_weights(const float* __restrict__ qw,
                             const float* __restrict__ kvw,
                             const float* __restrict__ pw,
                             h16* __restrict__ qwT,
                             h16* __restrict__ kvwT,
                             h16* __restrict__ pwT) {
  int idx = blockIdx.x * blockDim.x + threadIdx.x;
  if (idx < DIM * DIM) {
    int ci = idx / DIM, co = idx % DIM;
    qwT[idx] = (h16)qw[co * DIM + ci];
    pwT[idx] = (h16)pw[co * DIM + ci];
  }
  if (idx < DIM * KVDIM) {
    int ci = idx / KVDIM, co = idx % KVDIM;
    kvwT[idx] = (h16)kvw[co * DIM + ci];
  }
}

// ---------------- fused attention kernel ----------------
__global__ __launch_bounds__(256)
void xwin_attn_kernel(const float* __restrict__ x, const float* __restrict__ y,
                      const float* __restrict__ mask,
                      const float* __restrict__ rpb, const int* __restrict__ rel,
                      const h16* __restrict__ qwT, const float* __restrict__ qb,
                      const h16* __restrict__ kvwT, const float* __restrict__ kvb,
                      const h16* __restrict__ pwT, const float* __restrict__ pb,
                      float* __restrict__ out) {
  __shared__ h16   sX [NPAD * DIM];          // 12 KB  x (f16), rows padded w/ 0
  __shared__ h16   sY [NPAD * DIM];          // 12 KB  y (f16)
  __shared__ h16   sQ [NPAD * DIM];          // 12 KB  scaled Q  [token][chan]
  __shared__ h16   sKT[DIM * NPAD];          // 12 KB  K^T       [chan][token]
  __shared__ h16   sV [NPAD * DIM];          // 12 KB  V         [token][chan]
  __shared__ float sS [HEADS * NPAD * NPAD]; // 48 KB  scores (f32)
  __shared__ h16   sP [HEADS * NPAD * NPAD]; // 24 KB  softmax probs (f16)
  __shared__ h16   sO [NPAD * DIM];          // 12 KB  attn output

  const int w    = blockIdx.x;
  const int tid  = threadIdx.x;
  const int lane = tid & 31;
  const int wave = tid >> 5;

  // warm L2/L0 with the (tiny, shared) transposed weights
  {
    const char* wp0 = (const char*)qwT;
    const char* wp1 = (const char*)kvwT;
    const char* wp2 = (const char*)pwT;
    __builtin_prefetch(wp0 + tid * 72, 0, 3);   // 96*96*2   = 18432 B
    __builtin_prefetch(wp1 + tid * 144, 0, 3);  // 96*192*2  = 36864 B
    __builtin_prefetch(wp2 + tid * 72, 0, 3);
  }

  // ---- phase 0: stage x,y -> LDS f16 (16B chunks), zero pad rows 49..63 ----
  const float* xw = x + (size_t)w * NTOK * DIM;
  const float* yw = y + (size_t)w * NTOK * DIM;
  for (int c = tid; c < (NPAD * DIM) / 8; c += 256) {   // 768 chunks of 8 halves
    int r = c / (DIM / 8);
    int coff = (c - r * (DIM / 8)) * 8;
    v8h hx = {}, hy = {};
    if (r < NTOK) {
      const float4* px = (const float4*)(xw + r * DIM + coff);
      const float4* py = (const float4*)(yw + r * DIM + coff);
      float4 x0 = px[0], x1 = px[1], y0 = py[0], y1 = py[1];
      hx[0] = (h16)x0.x; hx[1] = (h16)x0.y; hx[2] = (h16)x0.z; hx[3] = (h16)x0.w;
      hx[4] = (h16)x1.x; hx[5] = (h16)x1.y; hx[6] = (h16)x1.z; hx[7] = (h16)x1.w;
      hy[0] = (h16)y0.x; hy[1] = (h16)y0.y; hy[2] = (h16)y0.z; hy[3] = (h16)y0.w;
      hy[4] = (h16)y1.x; hy[5] = (h16)y1.y; hy[6] = (h16)y1.z; hy[7] = (h16)y1.w;
    }
    *reinterpret_cast<v8h*>(sX + r * DIM + coff) = hx;
    *reinterpret_cast<v8h*>(sY + r * DIM + coff) = hy;
  }
  __syncthreads();

  // ---- phase 1: Q = x@qwT + qb (scaled); K^T, V = y@kvwT + kvb ----
  // 24 Q tiles + 24 K tiles (stored transposed) + 24 V tiles; 3 k-steps each.
  for (int t = wave; t < 72; t += 8) {
    v8f acc = {};
    if (t < 24) {                       // -------- Q --------
      int mT = t / 6, nT = t % 6;
#pragma unroll
      for (int ks = 0; ks < 3; ++ks) {
        v16h a = load_a_frag(sX, DIM, mT * 16, ks * 32, lane);
        v16h b = load_b_frag_vec(qwT, DIM, ks * 32, nT * 16, lane);
        acc = wmma_f16(a, b, acc);
      }
      int col = nT * 16 + (lane & 15);
      float bias = qb[col];
#pragma unroll
      for (int r = 0; r < 8; ++r) {
        int row = mT * 16 + r + ((lane >> 4) << 3);
        sQ[row * DIM + col] = (h16)((acc[r] + bias) * ATTN_SCALE);
      }
    } else if (t < 48) {                // -------- K (store transposed) --------
      int tt = t - 24;
      int mT = tt / 6, nT = tt % 6;     // nT over K-channels 0..95
#pragma unroll
      for (int ks = 0; ks < 3; ++ks) {
        v16h a = load_a_frag(sY, DIM, mT * 16, ks * 32, lane);
        v16h b = load_b_frag_vec(kvwT, KVDIM, ks * 32, nT * 16, lane);
        acc = wmma_f16(a, b, acc);
      }
      int col = nT * 16 + (lane & 15);  // channel
      float bias = kvb[col];
      v8h pk;                           // 8 consecutive token-rows of one chan
#pragma unroll
      for (int r = 0; r < 8; ++r) pk[r] = (h16)(acc[r] + bias);
      *reinterpret_cast<v8h*>(sKT + col * NPAD + mT * 16 + ((lane >> 4) << 3)) = pk;
    } else {                            // -------- V --------
      int tt = t - 48;
      int mT = tt / 6, nT = tt % 6;     // nT over V-channels 0..95
#pragma unroll
      for (int ks = 0; ks < 3; ++ks) {
        v16h a = load_a_frag(sY, DIM, mT * 16, ks * 32, lane);
        v16h b = load_b_frag_vec(kvwT, KVDIM, ks * 32, DIM + nT * 16, lane);
        acc = wmma_f16(a, b, acc);
      }
      int col = nT * 16 + (lane & 15);
      float bias = kvb[DIM + col];
#pragma unroll
      for (int r = 0; r < 8; ++r) {
        int row = mT * 16 + r + ((lane >> 4) << 3);
        sV[row * DIM + col] = (h16)(acc[r] + bias);
      }
    }
  }
  __syncthreads();

  // ---- phase 2: S = (scaled Q) @ K^T per head; K-dim = 32 = 1 WMMA ----
  // B-frag is a strided gather (stride NPAD) from sKT [chan][token].
  for (int t = wave; t < 48; t += 8) {
    int h  = t >> 4;
    int mT = (t & 15) >> 2;
    int nT = t & 3;
    v16h a = load_a_frag(sQ, DIM, mT * 16, h * HD, lane);
    v16h b = load_b_frag_vec(sKT, NPAD, h * HD, nT * 16, lane);
    v8f acc = {};
    acc = wmma_f16(a, b, acc);
    int col = nT * 16 + (lane & 15);
    float* dst = sS + h * NPAD * NPAD;
#pragma unroll
    for (int r = 0; r < 8; ++r) {
      int row = mT * 16 + r + ((lane >> 4) << 3);
      dst[row * NPAD + col] = acc[r];
    }
  }
  __syncthreads();

  // ---- phase 3: +rel-pos bias, +mask, row softmax -> sP (f16, zero-padded) ----
  const float* maskw = mask + (size_t)(w & 4095) * NTOK * NTOK;
  if (tid < HEADS * NPAD) {
    int h = tid / NPAD;
    int r = tid - h * NPAD;
    v8h* prow = reinterpret_cast<v8h*>(sP + (size_t)(h * NPAD + r) * NPAD);
    if (r < NTOK) {
      float* srow = sS + (size_t)(h * NPAD + r) * NPAD;
      const int*   rrow = rel + r * NTOK;
      const float* mrow = maskw + r * NTOK;
      float mx = -1e30f;
      for (int m = 0; m < NTOK; ++m) {
        float v = srow[m] + rpb[rrow[m] * HEADS + h] + mrow[m];
        srow[m] = v;
        mx = fmaxf(mx, v);
      }
      float sum = 0.f;
      for (int m = 0; m < NTOK; ++m) {
        float e = __expf(srow[m] - mx);
        srow[m] = e;
        sum += e;
      }
      float inv = 1.f / sum;
#pragma unroll
      for (int cc = 0; cc < 8; ++cc) {
        v8h pk = {};
#pragma unroll
        for (int j = 0; j < 8; ++j) {
          int m = cc * 8 + j;
          if (m < NTOK) pk[j] = (h16)(srow[m] * inv);
        }
        prow[cc] = pk;
      }
    } else {
      v8h z = {};
#pragma unroll
      for (int cc = 0; cc < 8; ++cc) prow[cc] = z;
    }
  }
  __syncthreads();

  // ---- phase 4: O = P @ V per head (K-dim 64 = 2 k-steps) ----
  for (int t = wave; t < 24; t += 8) {
    int h  = t / 8;
    int mT = (t & 7) >> 1;
    int nT = t & 1;
    const h16* ph = sP + (size_t)h * NPAD * NPAD;
    v8f acc = {};
#pragma unroll
    for (int ks = 0; ks < 2; ++ks) {
      v16h a = load_a_frag(ph, NPAD, mT * 16, ks * 32, lane);
      v16h b = load_b_frag_vec(sV, DIM, ks * 32, h * HD + nT * 16, lane);
      acc = wmma_f16(a, b, acc);
    }
    int col = h * HD + nT * 16 + (lane & 15);
#pragma unroll
    for (int r = 0; r < 8; ++r) {
      int row = mT * 16 + r + ((lane >> 4) << 3);
      sO[row * DIM + col] = (h16)acc[r];
    }
  }
  __syncthreads();

  // ---- phase 5: out = O @ pwT + pb  (f32 store, rows < 49) ----
  for (int t = wave; t < 24; t += 8) {
    int mT = t / 6, nT = t % 6;
    v8f acc = {};
#pragma unroll
    for (int ks = 0; ks < 3; ++ks) {
      v16h a = load_a_frag(sO, DIM, mT * 16, ks * 32, lane);
      v16h b = load_b_frag_vec(pwT, DIM, ks * 32, nT * 16, lane);
      acc = wmma_f16(a, b, acc);
    }
    int col = nT * 16 + (lane & 15);
    float bias = pb[col];
#pragma unroll
    for (int r = 0; r < 8; ++r) {
      int row = mT * 16 + r + ((lane >> 4) << 3);
      if (row < NTOK)
        out[((size_t)w * NTOK + row) * DIM + col] = acc[r] + bias;
    }
  }
}

extern "C" void kernel_launch(void* const* d_in, const int* in_sizes, int n_in,
                              void* d_out, int out_size, void* d_ws, size_t ws_size,
                              hipStream_t stream) {
  const float* x    = (const float*)d_in[0];
  const float* y    = (const float*)d_in[1];
  const float* mask = (const float*)d_in[2];
  const float* rpb  = (const float*)d_in[3];
  const int*   rel  = (const int*)  d_in[4];
  const float* qw   = (const float*)d_in[5];
  const float* qb   = (const float*)d_in[6];
  const float* kvw  = (const float*)d_in[7];
  const float* kvb  = (const float*)d_in[8];
  const float* pw   = (const float*)d_in[9];
  const float* pb   = (const float*)d_in[10];
  float* out = (float*)d_out;

  // workspace: transposed f16 weights (qwT | kvwT | pwT), ~72 KB
  h16* qwT  = (h16*)d_ws;
  h16* kvwT = qwT + DIM * DIM;
  h16* pwT  = kvwT + DIM * KVDIM;

  prep_weights<<<(DIM * KVDIM + 255) / 256, 256, 0, stream>>>(qw, kvw, pw,
                                                              qwT, kvwT, pwT);

  const int nWin = in_sizes[0] / (NTOK * DIM);   // 16384
  xwin_attn_kernel<<<nWin, 256, 0, stream>>>(x, y, mask, rpb, rel,
                                             qwT, qb, kvwT, kvb, pwT, pb, out);
}
